// BahdanauAttention_10900626997696
// MI455X (gfx1250) — matrix operand order repaired and checked by reading, not compiled, not run on hardware
//
#include <hip/hip_runtime.h>
#include <math.h>

#define Bn 64
#define Sn 1024
#define Hn 1024

typedef __attribute__((ext_vector_type(16))) __bf16 v16bf;
typedef __attribute__((ext_vector_type(8)))  __bf16 v8bf;
typedef __attribute__((ext_vector_type(4)))  __bf16 v4bf;
typedef __attribute__((ext_vector_type(8)))  float  v8f;

static __device__ __forceinline__ __bf16 f2bf(float f) { return (__bf16)f; }

// Branch-free tanh. gfx1250 has native v_tanh_f32; fall back to a
// branchless expf form with correct +/-1 limits if the builtin is absent.
static __device__ __forceinline__ float fast_tanh(float x) {
#if __has_builtin(__builtin_amdgcn_tanhf)
  return __builtin_amdgcn_tanhf(x);
#else
  float e = __expf(2.0f * x);          // inf for large x, 0 for very negative
  return 1.0f - 2.0f / (e + 1.0f);     // -> +1 / -1 at the limits, no NaN
#endif
}

// ---------------------------------------------------------------------------
// Kernel 1: convert W_h (fp32, row-major [k][h]) -> bf16, same layout.
// B-matrix column n (output k) == W_h row k, so B fragment loads are
// contiguous 16-element runs of one W_h row.
// ---------------------------------------------------------------------------
__global__ __launch_bounds__(256) void convert_wh_kernel(
    const float* __restrict__ wh, __bf16* __restrict__ whb) {
  int idx = (blockIdx.x * 256 + threadIdx.x) * 4;
  float4 f = *(const float4*)(wh + idx);
  v4bf o;
  o[0] = f2bf(f.x); o[1] = f2bf(f.y); o[2] = f2bf(f.z); o[3] = f2bf(f.w);
  *(v4bf*)(whb + idx) = o;
}

// ---------------------------------------------------------------------------
// Kernel 2: proj_s[b][k] = sum_h dec[b][h] * W_s[k][h]   (tiny: 0.13 GFLOP)
// ---------------------------------------------------------------------------
__global__ __launch_bounds__(256) void proj_s_kernel(
    const float* __restrict__ dec, const float* __restrict__ Ws,
    float* __restrict__ proj_s) {
  const int b = blockIdx.y;
  const int k = blockIdx.x * 256 + threadIdx.x;
  __shared__ float dsh[Hn];
  for (int i = threadIdx.x; i < Hn; i += 256) dsh[i] = dec[b * Hn + i];
  __syncthreads();
  const float* wr = Ws + (size_t)k * Hn;
  float acc = 0.f;
#pragma unroll 4
  for (int h = 0; h < Hn; h += 4) {
    float4 w = *(const float4*)(wr + h);
    acc += w.x * dsh[h] + w.y * dsh[h + 1] + w.z * dsh[h + 2] + w.w * dsh[h + 3];
  }
  proj_s[b * Hn + k] = acc;
}

// ---------------------------------------------------------------------------
// Kernel 3 (main): fused proj_h GEMM (bf16 WMMA) + tanh + dot(v) -> scores.
// Grid: (S/16, B). Block: 256 threads = 8 waves; wave w owns k-tiles
// [w*8, w*8+8), processed in PAIRS so each A fragment (LDS) feeds 2 WMMAs.
// LDS stages the 16x1024 enc tile as bf16 (stride 1040 keeps 16B alignment).
// ---------------------------------------------------------------------------
#define ASTRIDE 1040

__global__ __launch_bounds__(256) void scores_kernel(
    const float* __restrict__ enc, const __bf16* __restrict__ whb,
    const float* __restrict__ proj_s, const float* __restrict__ vvec,
    const int* __restrict__ mask, float* __restrict__ scores) {
  const int b   = blockIdx.y;
  const int s0  = blockIdx.x * 16;
  const int tid = threadIdx.x;
  const int wave = tid >> 5;
  const int lane = tid & 31;
  const int half = lane >> 4;   // 0: lanes 0-15, 1: lanes 16-31
  const int l16  = lane & 15;

  __shared__ __align__(16) __bf16 As[16 * ASTRIDE];
  __shared__ float sscore[16];

  if (tid < 16) sscore[tid] = 0.f;

  // Stage enc[b, s0:s0+16, :] (contiguous 64KB) -> bf16 LDS tile.
  const float4* src4 = (const float4*)(enc + ((size_t)b * Sn + s0) * Hn);
  for (int i = tid; i < (16 * Hn) / 4; i += 256) {
    int e = i * 4;
    int row = e >> 10;
    int col = e & (Hn - 1);
    float4 f = src4[i];
    v4bf o;
    o[0] = f2bf(f.x); o[1] = f2bf(f.y); o[2] = f2bf(f.z); o[3] = f2bf(f.w);
    *(v4bf*)(&As[row * ASTRIDE + col]) = o;
  }
  __syncthreads();

  const __bf16* Arow = As + l16 * ASTRIDE;

  float part[8];
#pragma unroll
  for (int r = 0; r < 8; ++r) part[r] = 0.f;

  for (int ktp = 0; ktp < 4; ++ktp) {
    const int kbase0 = (wave * 8 + ktp * 2) * 16;      // first tile of pair
    // B fragment base: lane holds column N = l16 of W_h row (kbase + l16)
    const __bf16* B0 = whb + (size_t)(kbase0 + l16) * Hn;
    const __bf16* B1 = B0 + (size_t)16 * Hn;

    v8f c0 = {};
    v8f c1 = {};
#pragma unroll 4
    for (int kk = 0; kk < Hn; kk += 32) {
      // A fragment (contiguous k order, same as B): e0-15 -> K = half*16 + 0..15.
      // Loaded once, used twice.
      v8bf a0 = *(const v8bf*)(Arow + kk + half * 16);
      v8bf a1 = *(const v8bf*)(Arow + kk + half * 16 + 8);
      v16bf Af = __builtin_shufflevector(a0, a1, 0, 1, 2, 3, 4, 5, 6, 7,
                                         8, 9, 10, 11, 12, 13, 14, 15);
      // B fragments (32x16): K = half*16 + e
      v8bf b00 = *(const v8bf*)(B0 + kk + half * 16);
      v8bf b01 = *(const v8bf*)(B0 + kk + half * 16 + 8);
      v8bf b10 = *(const v8bf*)(B1 + kk + half * 16);
      v8bf b11 = *(const v8bf*)(B1 + kk + half * 16 + 8);
      v16bf Bf0 = __builtin_shufflevector(b00, b01, 0, 1, 2, 3, 4, 5, 6, 7,
                                          8, 9, 10, 11, 12, 13, 14, 15);
      v16bf Bf1 = __builtin_shufflevector(b10, b11, 0, 1, 2, 3, 4, 5, 6, 7,
                                          8, 9, 10, 11, 12, 13, 14, 15);
      c0 = __builtin_amdgcn_wmma_f32_16x16x32_bf16(
               false, Af, false, Bf0, (short)0, c0, false, false);
      c1 = __builtin_amdgcn_wmma_f32_16x16x32_bf16(
               false, Af, false, Bf1, (short)0, c1, false, false);
    }

    // C/D layout: c[r] = proj_h[s0 + r + half*8][kbase + l16]
    const int kg0 = kbase0 + l16;
    const int kg1 = kg0 + 16;
    const float ps0 = proj_s[b * Hn + kg0];
    const float ps1 = proj_s[b * Hn + kg1];
    const float vk0 = vvec[kg0];
    const float vk1 = vvec[kg1];
#pragma unroll
    for (int r = 0; r < 8; ++r) {
      part[r] += fast_tanh(c0[r] + ps0) * vk0;
      part[r] += fast_tanh(c1[r] + ps1) * vk1;
    }
  }

  // Reduce over the 16 lanes of each half (k dimension within this wave).
#pragma unroll
  for (int r = 0; r < 8; ++r) {
    for (int off = 8; off >= 1; off >>= 1)
      part[r] += __shfl_xor(part[r], off, 32);
  }
  if (l16 == 0) {
#pragma unroll
    for (int r = 0; r < 8; ++r)
      atomicAdd(&sscore[half * 8 + r], part[r]);   // ds_add_f32
  }
  __syncthreads();

  if (tid < 16) {
    int s = s0 + tid;
    float sc = sscore[tid];
    if (mask[b * Sn + s] == 0) sc = -__builtin_inff();
    scores[b * Sn + s] = sc;
  }
}

// ---------------------------------------------------------------------------
// Kernel 4: row softmax over S per batch row.
// ---------------------------------------------------------------------------
__global__ __launch_bounds__(256) void softmax_kernel(
    const float* __restrict__ scores, float* __restrict__ weights) {
  const int b = blockIdx.x;
  const int tid = threadIdx.x;
  __shared__ float red[256];

  float m = -__builtin_inff();
  for (int s = tid; s < Sn; s += 256) m = fmaxf(m, scores[b * Sn + s]);
  red[tid] = m;
  __syncthreads();
  for (int o = 128; o > 0; o >>= 1) {
    if (tid < o) red[tid] = fmaxf(red[tid], red[tid + o]);
    __syncthreads();
  }
  const float mx = red[0];
  __syncthreads();

  float sum = 0.f;
  for (int s = tid; s < Sn; s += 256) {
    float e = __expf(scores[b * Sn + s] - mx);
    weights[b * Sn + s] = e;
    sum += e;
  }
  red[tid] = sum;
  __syncthreads();
  for (int o = 128; o > 0; o >>= 1) {
    if (tid < o) red[tid] += red[tid + o];
    __syncthreads();
  }
  const float inv = 1.f / red[0];
  __syncthreads();
  for (int s = tid; s < Sn; s += 256) weights[b * Sn + s] *= inv;
}

// ---------------------------------------------------------------------------
// Kernel 5: context[b][h] = sum_s weights[b][s] * enc[b][s][h]  (BW bound)
// ---------------------------------------------------------------------------
__global__ __launch_bounds__(256) void context_kernel(
    const float* __restrict__ enc, const float* __restrict__ weights,
    float* __restrict__ ctx) {
  const int b = blockIdx.y;
  const int h = blockIdx.x * 256 + threadIdx.x;
  const float* e = enc + (size_t)b * Sn * Hn + h;
  const float* w = weights + b * Sn;
  float acc = 0.f;
#pragma unroll 4
  for (int s = 0; s < Sn; ++s) acc += w[s] * e[(size_t)s * Hn];
  ctx[b * Hn + h] = acc;
}

// ---------------------------------------------------------------------------
extern "C" void kernel_launch(void* const* d_in, const int* in_sizes, int n_in,
                              void* d_out, int out_size, void* d_ws, size_t ws_size,
                              hipStream_t stream) {
  const float* dec  = (const float*)d_in[0];   // [B,H]
  const float* enc  = (const float*)d_in[1];   // [B,S,H]
  const int*   mask = (const int*)d_in[2];     // [B,S]
  const float* Ws   = (const float*)d_in[3];   // [H,H]
  const float* Wh   = (const float*)d_in[4];   // [H,H]
  const float* v    = (const float*)d_in[5];   // [H]

  float* ctx_out = (float*)d_out;              // [B,H]
  float* w_out   = (float*)d_out + Bn * Hn;    // [B,S]

  char* ws = (char*)d_ws;
  __bf16* whb    = (__bf16*)ws;                          // H*H*2  = 2 MB
  float*  projs  = (float*)(ws + (size_t)Hn * Hn * 2);   // B*H*4  = 256 KB
  float*  scores = projs + Bn * Hn;                      // B*S*4  = 256 KB

  convert_wh_kernel<<<(Hn * Hn) / (256 * 4), 256, 0, stream>>>(Wh, whb);
  proj_s_kernel<<<dim3(Hn / 256, Bn), 256, 0, stream>>>(dec, Ws, projs);
  scores_kernel<<<dim3(Sn / 16, Bn), 256, 0, stream>>>(enc, whb, projs, v,
                                                       mask, scores);
  softmax_kernel<<<Bn, 256, 0, stream>>>(scores, w_out);
  context_kernel<<<dim3(Hn / 256, Bn), 256, 0, stream>>>(enc, w_out, ctx_out);
}
